// FusedMoEModularKernel_84705345012305
// MI455X (gfx1250) — matrix operand order repaired and checked by reading, not compiled, not run on hardware
//
#include <hip/hip_runtime.h>
#include <stdint.h>

typedef unsigned short ushort_t;
typedef __attribute__((ext_vector_type(16))) __bf16 v16bf;
typedef __attribute__((ext_vector_type(8)))  float  v8f;

// Problem constants (match reference)
#define MOE_E     8
#define MOE_TOPK  2
#define MOE_M     8192
#define MOE_K     2048
#define MOE_N     1408
#define MOE_CAP   (MOE_M * MOE_TOPK)          // 16384 routed (token,weight) entries
#define ROWS_PAD  (MOE_CAP + MOE_E * 128)     // per-expert segments padded to 128
#define MAX_TILES (ROWS_PAD / 128)            // 136
#define BM        128                         // token rows per block
#define BK        32                          // k-chunk (one WMMA K)
#define THREADS   256                         // 8 waves (wave32)

struct U32x8 { uint4 lo, hi; };

static __device__ inline v16bf as_frag(uint4 lo, uint4 hi) {
    U32x8 t; t.lo = lo; t.hi = hi;
    return __builtin_bit_cast(v16bf, t);
}

static __device__ inline ushort_t f32_to_bf16(float f) {
    union { float f; uint32_t u; } v; v.f = f;
    uint32_t r = v.u + 0x7FFFu + ((v.u >> 16) & 1u);   // round-to-nearest-even
    return (ushort_t)(r >> 16);
}

// LDS flat-pointer -> 32-bit LDS offset (aperture rule: low 32 bits of the address)
static __device__ inline unsigned lds_off32(const void* p) {
    return (unsigned)(uintptr_t)p;
}

// ASYNCcnt-tracked global->LDS b128 copy (gfx1250 async data path)
static __device__ inline void async_ld_b128(unsigned lds_byte_off, const void* gaddr) {
    asm volatile("global_load_async_to_lds_b128 %0, %1, off"
                 :: "v"(lds_byte_off), "v"((unsigned long long)gaddr)
                 : "memory");
}
static __device__ inline void wait_async0() {
    asm volatile("s_wait_asynccnt 0" ::: "memory");
}

// ---------------------------------------------------------------------------
// fp32 -> bf16 bulk convert (grid-stride)
// ---------------------------------------------------------------------------
__global__ void cvt_f32_bf16_kernel(const float* __restrict__ src,
                                    ushort_t* __restrict__ dst, int n) {
    int stride = gridDim.x * blockDim.x;
    for (int i = blockIdx.x * blockDim.x + threadIdx.x; i < n; i += stride)
        dst[i] = f32_to_bf16(src[i]);
}

// ---------------------------------------------------------------------------
// Routing: histogram -> tiny scan -> scatter into padded per-expert segments
// meta layout (ints): [0..7]=cnt, [8..15]=tileBase, [16..23]=tileCnt, [24..31]=cursor
// ---------------------------------------------------------------------------
__global__ void hist_kernel(const int* __restrict__ topk_ids, int* __restrict__ meta) {
    int i = blockIdx.x * blockDim.x + threadIdx.x;
    if (i >= MOE_M * MOE_TOPK) return;
    atomicAdd(&meta[topk_ids[i]], 1);
}

__global__ void scan_kernel(int* __restrict__ meta) {
    if (threadIdx.x != 0 || blockIdx.x != 0) return;
    int total = 0;
    for (int e = 0; e < MOE_E; ++e) {
        int tiles = (meta[e] + BM - 1) / BM;
        meta[8 + e]  = total;   // tileBase
        meta[16 + e] = tiles;   // tileCnt
        meta[24 + e] = 0;       // cursor
        total += tiles;
    }
}

__global__ void scatter_kernel(const int* __restrict__ topk_ids,
                               const float* __restrict__ topk_w,
                               int* __restrict__ meta,
                               int* __restrict__ tok,
                               float* __restrict__ wgt) {
    int i = blockIdx.x * blockDim.x + threadIdx.x;
    if (i >= MOE_M * MOE_TOPK) return;
    int t = i / MOE_TOPK;
    int e = topk_ids[i];
    int pos = atomicAdd(&meta[24 + e], 1);
    int row = meta[8 + e] * BM + pos;
    tok[row] = t;
    wgt[row] = topk_w[i];
}

// map a global 128-row tile index -> expert (or -1)
static __device__ inline int tile_to_expert(const int* meta, int gt) {
    int e = -1;
    #pragma unroll
    for (int i = 0; i < MOE_E; ++i) {
        int tb = meta[8 + i], tc = meta[16 + i];
        if (gt >= tb && gt < tb + tc) e = i;
    }
    return e;
}

// ---------------------------------------------------------------------------
// GEMM1: h[row, n] = silu(x@Wg^T) * (x@Wu^T)    (128 rows x 64 cols per block)
// wave layout: wave = wr(2) x wc(4); wave computes 4 row-subtiles x 16 cols
// double-buffered async global->LDS staging of the A (token) tile
// ---------------------------------------------------------------------------
__global__ __launch_bounds__(THREADS)
void gemm1_kernel(const ushort_t* __restrict__ xb,    // [M,K] bf16
                  const ushort_t* __restrict__ w1b,   // [E,2N,K] bf16
                  const int*      __restrict__ meta,
                  const int*      __restrict__ tok,   // [ROWS_PAD]
                  ushort_t*       __restrict__ hb) {  // [ROWS_PAD,N] bf16
    __shared__ __align__(16) ushort_t lds_a[2][BM * BK];  // 2 x 8 KB
    __shared__ int lds_tok[BM];

    const int gt = blockIdx.x;
    const int e  = tile_to_expert(meta, gt);
    if (e < 0) return;
    const int row0 = gt * BM;

    const int tid  = threadIdx.x;
    const int lane = tid & 31;
    const int wave = tid >> 5;
    const int wr   = wave & 1;        // row half (0..1)
    const int wc   = wave >> 1;       // col subtile (0..3)
    const int col  = lane & 15;
    const int ksel = lane >> 4;

    if (tid < BM) lds_tok[tid] = tok[row0 + tid];
    __syncthreads();

    // ---- per-thread fixed staging slots: 2 b128 vectors per chunk ----
    const int sr0 = tid >> 2;                  // rows 0..63
    const int sr1 = sr0 + 64;                  // rows 64..127
    const int sc  = (tid & 3) * 8;             // col offset (elements)
    const int t0 = lds_tok[sr0];
    const int t1 = lds_tok[sr1];
    const ushort_t* gp0 = xb + (size_t)(t0 < 0 ? 0 : t0) * MOE_K + sc;
    const ushort_t* gp1 = xb + (size_t)(t1 < 0 ? 0 : t1) * MOE_K + sc;
    unsigned d0[2], d1[2];
    #pragma unroll
    for (int b = 0; b < 2; ++b) {
        d0[b] = lds_off32(&lds_a[b][sr0 * BK + sc]);
        d1[b] = lds_off32(&lds_a[b][sr1 * BK + sc]);
    }
    // zero padding rows once (both buffers); never re-staged
    uint4 z4 = make_uint4(0u, 0u, 0u, 0u);
    if (t0 < 0) { *(uint4*)&lds_a[0][sr0 * BK + sc] = z4; *(uint4*)&lds_a[1][sr0 * BK + sc] = z4; }
    if (t1 < 0) { *(uint4*)&lds_a[0][sr1 * BK + sc] = z4; *(uint4*)&lds_a[1][sr1 * BK + sc] = z4; }

    const int n0 = blockIdx.y * 64 + wc * 16;
    const ushort_t* w1e = w1b + (size_t)e * (2 * MOE_N) * MOE_K;
    const ushort_t* bg_row = w1e + (size_t)(n0 + col) * MOE_K;
    const ushort_t* bu_row = w1e + (size_t)(MOE_N + n0 + col) * MOE_K;

    v8f accg[4], accu[4];
    #pragma unroll
    for (int s = 0; s < 4; ++s) { accg[s] = (v8f){}; accu[s] = (v8f){}; }

    // prologue: stage chunk 0 into buffer 0
    if (t0 >= 0) async_ld_b128(d0[0], gp0);
    if (t1 >= 0) async_ld_b128(d1[0], gp1);
    wait_async0();
    __syncthreads();

    int buf = 0;
    for (int kb = 0; kb < MOE_K; kb += BK, buf ^= 1) {
        // stage next chunk into the other buffer (overlaps with WMMA below)
        if (kb + BK < MOE_K) {
            if (t0 >= 0) async_ld_b128(d0[buf ^ 1], gp0 + kb + BK);
            if (t1 >= 0) async_ld_b128(d1[buf ^ 1], gp1 + kb + BK);
        }

        // per-wave B fragments (gate + up), 32x16 bf16 each
        int kbB = kb + ksel * 8;
        uint4 g0 = *(const uint4*)(bg_row + kbB);
        uint4 g1 = *(const uint4*)(bg_row + kbB + 16);
        uint4 u0 = *(const uint4*)(bu_row + kbB);
        uint4 u1 = *(const uint4*)(bu_row + kbB + 16);
        __builtin_prefetch(bg_row + kbB + 64, 0, 1);
        __builtin_prefetch(bu_row + kbB + 64, 0, 1);
        v16bf bG = as_frag(g0, g1);
        v16bf bU = as_frag(u0, u1);

        // 4 row-subtiles x (gate,up) = 8 WMMAs sharing B
        const ushort_t* ab = lds_a[buf];
        #pragma unroll
        for (int s = 0; s < 4; ++s) {
            int r = wr * 64 + s * 16 + col;
            int o0 = ksel * 8;
            uint4 alo = *(const uint4*)&ab[r * BK + o0];
            uint4 ahi = *(const uint4*)&ab[r * BK + o0 + 16];
            v16bf a = as_frag(alo, ahi);
            accg[s] = __builtin_amdgcn_wmma_f32_16x16x32_bf16(
                          false, a, false, bG, (short)0, accg[s], false, false);
            accu[s] = __builtin_amdgcn_wmma_f32_16x16x32_bf16(
                          false, a, false, bU, (short)0, accu[s], false, false);
        }
        wait_async0();
        __syncthreads();
    }

    // silu(g)*u -> bf16 h rows
    #pragma unroll
    for (int s = 0; s < 4; ++s) {
        #pragma unroll
        for (int i = 0; i < 8; ++i) {
            float g = accg[s][i], u = accu[s][i];
            float h = (g / (1.0f + __expf(-g))) * u;
            int mr = wr * 64 + s * 16 + i + 8 * ksel;   // C layout: VGPR i -> row i / i+8
            hb[(size_t)(row0 + mr) * MOE_N + n0 + col] = f32_to_bf16(h);
        }
    }
}

// ---------------------------------------------------------------------------
// GEMM2: out[t, k] += coef * (h @ w2^T)     (128 rows x 64 cols per block)
// ---------------------------------------------------------------------------
__global__ __launch_bounds__(THREADS)
void gemm2_kernel(const ushort_t* __restrict__ hb,    // [ROWS_PAD,N] bf16
                  const ushort_t* __restrict__ w2b,   // [E,K,N] bf16
                  const int*      __restrict__ meta,
                  const int*      __restrict__ tok,
                  const float*    __restrict__ wgt,
                  float*          __restrict__ out) { // [M,K] f32
    __shared__ __align__(16) ushort_t lds_a[2][BM * BK];  // 2 x 8 KB
    __shared__ int   lds_tok[BM];
    __shared__ float lds_coef[BM];

    const int gt = blockIdx.x;
    const int e  = tile_to_expert(meta, gt);
    if (e < 0) return;
    const int row0 = gt * BM;

    const int tid  = threadIdx.x;
    const int lane = tid & 31;
    const int wave = tid >> 5;
    const int wr   = wave & 1;
    const int wc   = wave >> 1;
    const int col  = lane & 15;
    const int ksel = lane >> 4;

    if (tid < BM) { lds_tok[tid] = tok[row0 + tid]; lds_coef[tid] = wgt[row0 + tid]; }

    // per-thread fixed staging slots (h rows are contiguous & always valid)
    const int sr0 = tid >> 2;
    const int sr1 = sr0 + 64;
    const int sc  = (tid & 3) * 8;
    const ushort_t* gp0 = hb + (size_t)(row0 + sr0) * MOE_N + sc;
    const ushort_t* gp1 = hb + (size_t)(row0 + sr1) * MOE_N + sc;
    unsigned d0[2], d1[2];
    #pragma unroll
    for (int b = 0; b < 2; ++b) {
        d0[b] = lds_off32(&lds_a[b][sr0 * BK + sc]);
        d1[b] = lds_off32(&lds_a[b][sr1 * BK + sc]);
    }

    const int k0 = blockIdx.y * 64 + wc * 16;
    const ushort_t* w2e = w2b + (size_t)e * MOE_K * MOE_N;
    const ushort_t* b_row = w2e + (size_t)(k0 + col) * MOE_N;

    v8f acc[4];
    #pragma unroll
    for (int s = 0; s < 4; ++s) acc[s] = (v8f){};

    // prologue: stage chunk 0 into buffer 0
    async_ld_b128(d0[0], gp0);
    async_ld_b128(d1[0], gp1);
    wait_async0();
    __syncthreads();

    int buf = 0;
    for (int nb = 0; nb < MOE_N; nb += BK, buf ^= 1) {
        if (nb + BK < MOE_N) {
            async_ld_b128(d0[buf ^ 1], gp0 + nb + BK);
            async_ld_b128(d1[buf ^ 1], gp1 + nb + BK);
        }

        int nbB = nb + ksel * 8;
        uint4 b0 = *(const uint4*)(b_row + nbB);
        uint4 b1 = *(const uint4*)(b_row + nbB + 16);
        __builtin_prefetch(b_row + nbB + 64, 0, 1);
        v16bf b = as_frag(b0, b1);

        const ushort_t* ab = lds_a[buf];
        #pragma unroll
        for (int s = 0; s < 4; ++s) {
            int r = wr * 64 + s * 16 + col;
            int o0 = ksel * 8;
            uint4 alo = *(const uint4*)&ab[r * BK + o0];
            uint4 ahi = *(const uint4*)&ab[r * BK + o0 + 16];
            v16bf a = as_frag(alo, ahi);
            acc[s] = __builtin_amdgcn_wmma_f32_16x16x32_bf16(
                         false, a, false, b, (short)0, acc[s], false, false);
        }
        wait_async0();
        __syncthreads();
    }

    // scale by routing coef, atomic-accumulate into out
    #pragma unroll
    for (int s = 0; s < 4; ++s) {
        #pragma unroll
        for (int i = 0; i < 8; ++i) {
            int mr = wr * 64 + s * 16 + i + 8 * ksel;
            int t = lds_tok[mr];
            if (t >= 0)
                atomicAdd(&out[(size_t)t * MOE_K + k0 + col], acc[s][i] * lds_coef[mr]);
        }
    }
}

// ---------------------------------------------------------------------------
// Launcher
// ---------------------------------------------------------------------------
extern "C" void kernel_launch(void* const* d_in, const int* in_sizes, int n_in,
                              void* d_out, int out_size, void* d_ws, size_t ws_size,
                              hipStream_t stream) {
    const float* x       = (const float*)d_in[0];   // [M,K]
    const float* w1      = (const float*)d_in[1];   // [E,2N,K]
    const float* w2      = (const float*)d_in[2];   // [E,K,N]
    const int*   topk_id = (const int*)  d_in[3];   // [M,TOPK]
    const float* topk_w  = (const float*)d_in[4];   // [M,TOPK]
    float* out = (float*)d_out;

    // workspace layout (bytes, 256-aligned)
    char* ws = (char*)d_ws;
    size_t off = 0;
    int*      meta = (int*)(ws + off);      off += 256;
    int*      tok  = (int*)(ws + off);      off += (size_t)ROWS_PAD * sizeof(int);
    float*    wgt  = (float*)(ws + off);    off += (size_t)ROWS_PAD * sizeof(float);
    off = (off + 255) & ~(size_t)255;
    ushort_t* xb   = (ushort_t*)(ws + off); off += (size_t)MOE_M * MOE_K * 2;
    ushort_t* w1b  = (ushort_t*)(ws + off); off += (size_t)MOE_E * 2 * MOE_N * MOE_K * 2;
    ushort_t* w2b  = (ushort_t*)(ws + off); off += (size_t)MOE_E * MOE_K * MOE_N * 2;
    ushort_t* hb   = (ushort_t*)(ws + off); off += (size_t)ROWS_PAD * MOE_N * 2;
    (void)ws_size; (void)n_in; (void)in_sizes;

    // init (graph-capture safe)
    hipMemsetAsync(meta, 0, 256, stream);
    hipMemsetAsync(tok, 0xFF, (size_t)ROWS_PAD * sizeof(int), stream);   // -1
    hipMemsetAsync(wgt, 0,    (size_t)ROWS_PAD * sizeof(float), stream);
    hipMemsetAsync(out, 0,    (size_t)out_size * sizeof(float), stream);

    // fp32 -> bf16 staging
    cvt_f32_bf16_kernel<<<2048, 256, 0, stream>>>(x,  xb,  MOE_M * MOE_K);
    cvt_f32_bf16_kernel<<<2048, 256, 0, stream>>>(w1, w1b, MOE_E * 2 * MOE_N * MOE_K);
    cvt_f32_bf16_kernel<<<2048, 256, 0, stream>>>(w2, w2b, MOE_E * MOE_K * MOE_N);

    // routing
    const int nent = MOE_M * MOE_TOPK;
    hist_kernel<<<(nent + 255) / 256, 256, 0, stream>>>(topk_id, meta);
    scan_kernel<<<1, 1, 0, stream>>>(meta);
    scatter_kernel<<<(nent + 255) / 256, 256, 0, stream>>>(topk_id, topk_w, meta, tok, wgt);

    // GEMM1: x @ w1^T -> silu*mul -> h
    gemm1_kernel<<<dim3(MAX_TILES, MOE_N / 64), THREADS, 0, stream>>>(
        xb, w1b, meta, tok, hb);

    // GEMM2: h @ w2^T -> weighted atomic accumulate into out
    gemm2_kernel<<<dim3(MAX_TILES, MOE_K / 64), THREADS, 0, stream>>>(
        hb, w2b, meta, tok, wgt, out);
}
